// CaptionAttentionC_37108517438105
// MI455X (gfx1250) — compile-verified, matrix operand order checked by reading, not run on hardware
//
#include <hip/hip_runtime.h>
#include <cstddef>

#define B_  32
#define L_  1024
#define DC_ 1024
#define DD_ 1024
#define A_  1024
#define NEGV -1e10f

typedef __attribute__((ext_vector_type(16))) _Float16 v16h;
typedef __attribute__((ext_vector_type(8)))  float    v8f;

constexpr int KTILES = DC_ / 32;  // 32 k-tiles of 32
constexpr int NTILES = A_ / 16;   // 64 n-tiles of 16

__device__ inline float wave_reduce_sum(float v) {
    v += __shfl_xor(v, 1, 32);
    v += __shfl_xor(v, 2, 32);
    v += __shfl_xor(v, 4, 32);
    v += __shfl_xor(v, 8, 32);
    v += __shfl_xor(v, 16, 32);
    return v;
}

// ---------------------------------------------------------------- Wf -> f16
__global__ __launch_bounds__(256)
void conv_wf_kernel(const float* __restrict__ Wf, _Float16* __restrict__ wf16, int n) {
    int i = blockIdx.x * 256 + threadIdx.x;
    if (i < n) wf16[i] = (_Float16)Wf[i];
}

// ------------------------------------------- u[b][a] = dh[b].Wd[a] + bd + bf
__global__ __launch_bounds__(256)
void u_kernel(const float* __restrict__ dh, const float* __restrict__ Wd,
              const float* __restrict__ bd, const float* __restrict__ bf,
              float* __restrict__ u) {
    const int lane = threadIdx.x & 31;
    const int wid  = blockIdx.x * 8 + (threadIdx.x >> 5);  // 0 .. B*A-1
    const int b = wid >> 10;
    const int a = wid & (A_ - 1);
    const float* __restrict__ x = dh + (size_t)b * DD_;
    const float* __restrict__ w = Wd + (size_t)a * DD_;
    float s = 0.f;
    for (int k = lane; k < DD_; k += 32) s += x[k] * w[k];
    s = wave_reduce_sum(s);
    if (lane == 0) u[wid] = s + bd[a] + bf[a];
}

// ------- fused: scores[b][l] = sum_a tanh(cap@Wf^T + u) * Wa[a]  (WMMA f16)
__global__ __launch_bounds__(256)
void scores_kernel(const float* __restrict__ cap, const _Float16* __restrict__ wf16,
                   const float* __restrict__ u, const float* __restrict__ Wa,
                   const float* __restrict__ ba, float* __restrict__ scores) {
    __shared__ _Float16 tileB[16][DC_ + 8];   // +8 f16 pad -> conflict-free

    const int lane   = threadIdx.x & 31;
    const int wave   = threadIdx.x >> 5;
    const int tile   = blockIdx.x * 8 + wave; // 16-row M tile id (0..2047)
    const int m_base = tile * 16;
    const int b      = m_base / L_;           // tiles never straddle batches
    const int koff   = (lane < 16) ? 0 : 8;   // K sub-offset per ISA layout
    const int ncol   = lane & 15;

    // --- stage this wave's 16x1024 A-slice into VGPRs as f16 (read cap once)
    const float* __restrict__ rowp = cap + (size_t)(m_base + ncol) * DC_;
    v16h a_frag[KTILES];
#pragma unroll
    for (int kt = 0; kt < KTILES; ++kt) {
        const float* p = rowp + kt * 32 + koff;
#pragma unroll
        for (int i = 0; i < 8; ++i) {
            a_frag[kt][i]     = (_Float16)p[i];
            a_frag[kt][8 + i] = (_Float16)p[16 + i];
        }
    }

    float partial[8];
#pragma unroll
    for (int r = 0; r < 8; ++r) partial[r] = 0.f;

    for (int nt = 0; nt < NTILES; ++nt) {
        const int n_base = nt * 16;
        __syncthreads();
        {   // cooperative LDS stage of Wf^T tile: 16 cols (n) x 1024 (k), f16
            const int r = threadIdx.x >> 4;        // 0..15  (n row of tile)
            const int c = (threadIdx.x & 15) * 64; // 64-elem chunk
            const float4* src = (const float4*)(wf16 + (size_t)(n_base + r) * DC_ + c);
            float4* dst = (float4*)(&tileB[r][c]);
#pragma unroll
            for (int i = 0; i < 8; ++i) dst[i] = src[i];
        }
        __syncthreads();

        v8f acc = {};
        const _Float16* __restrict__ bp = &tileB[ncol][0];
#pragma unroll
        for (int kt = 0; kt < KTILES; ++kt) {
            v16h bfr;
            const _Float16* q = bp + kt * 32 + koff;
#pragma unroll
            for (int i = 0; i < 8; ++i) { bfr[i] = q[i]; bfr[8 + i] = q[16 + i]; }
            acc = __builtin_amdgcn_wmma_f32_16x16x32_f16(
                      false, a_frag[kt], false, bfr, (short)0, acc, false, false);
        }
        // epilogue: C tile lane layout -> N = lane&15, M = r + (lane<16?0:8)
        const float uval = u[(size_t)b * A_ + n_base + ncol];
        const float wav  = Wa[n_base + ncol];
#pragma unroll
        for (int r = 0; r < 8; ++r)
            partial[r] += tanhf(acc[r] + uval) * wav;
    }

    // reduce over the 16 N-lanes of each half-wave
#pragma unroll
    for (int r = 0; r < 8; ++r) {
        float v = partial[r];
        v += __shfl_xor(v, 1, 32);
        v += __shfl_xor(v, 2, 32);
        v += __shfl_xor(v, 4, 32);
        v += __shfl_xor(v, 8, 32);
        partial[r] = v;
    }
    if ((lane & 15) == 0) {
        const float bav  = ba[0];
        const int   mrow = m_base + ((lane >> 4) << 3);  // +0 or +8
#pragma unroll
        for (int r = 0; r < 8; ++r)
            scores[mrow + r] = partial[r] + bav;
    }
}

// ---------------------------------------------------- masked softmax over L
__global__ __launch_bounds__(256)
void softmax_kernel(const float* __restrict__ scores, const int* __restrict__ mask,
                    float* __restrict__ alpha) {
    __shared__ float red[256];
    const int b = blockIdx.x, t = threadIdx.x;
    float v[4];
    float mx = -3.0e38f;
#pragma unroll
    for (int i = 0; i < 4; ++i) {
        const int l = t + i * 256;
        float s = scores[b * L_ + l];
        s = (mask[b * L_ + l] == 0) ? NEGV : s;
        v[i] = s;
        mx = fmaxf(mx, s);
    }
    red[t] = mx; __syncthreads();
    for (int o = 128; o > 0; o >>= 1) { if (t < o) red[t] = fmaxf(red[t], red[t + o]); __syncthreads(); }
    mx = red[0]; __syncthreads();
    float sm = 0.f;
#pragma unroll
    for (int i = 0; i < 4; ++i) { v[i] = __expf(v[i] - mx); sm += v[i]; }
    red[t] = sm; __syncthreads();
    for (int o = 128; o > 0; o >>= 1) { if (t < o) red[t] += red[t + o]; __syncthreads(); }
    const float inv = 1.f / red[0];
#pragma unroll
    for (int i = 0; i < 4; ++i) alpha[b * L_ + t + i * 256] = v[i] * inv;
}

// ---------------------------------------------- context[b][d] = alpha . cap
__global__ __launch_bounds__(256)
void context_kernel(const float* __restrict__ alpha, const float* __restrict__ cap,
                    float* __restrict__ ctx) {
    const int b = blockIdx.y;
    const int d = blockIdx.x * 256 + threadIdx.x;
    const float* __restrict__ ar = alpha + b * L_;
    const float* __restrict__ cr = cap + (size_t)b * L_ * DC_ + d;
    float s = 0.f;
    for (int l = 0; l < L_; ++l) s += ar[l] * cr[(size_t)l * DC_];
    ctx[b * DC_ + d] = s;
}

// ------------------------- gated fusion: out = zt*sc + (1-zt)*tc, wave/output
__global__ __launch_bounds__(256)
void fuse_kernel(const float* __restrict__ word, const float* __restrict__ dh,
                 const float* __restrict__ ctx,  const float* __restrict__ Wg,
                 const float* __restrict__ bg,   const float* __restrict__ Ws,
                 const float* __restrict__ bs,   const float* __restrict__ Wt,
                 const float* __restrict__ bt,   float* __restrict__ out) {
    const int lane = threadIdx.x & 31;
    const int wid  = blockIdx.x * 8 + (threadIdx.x >> 5);   // 0 .. B*DC-1
    const int b = wid >> 10;
    const int d = wid & (DC_ - 1);
    const float* __restrict__ xw = word + (size_t)b * DC_;
    const float* __restrict__ xh = dh   + (size_t)b * DD_;
    const float* __restrict__ xc = ctx  + (size_t)b * DC_;
    const float* __restrict__ wg = Wg + (size_t)d * (2 * DC_ + DD_);
    const float* __restrict__ ws = Ws + (size_t)d * DC_;
    const float* __restrict__ wt = Wt + (size_t)d * (DC_ + DD_);
    float z = 0.f, s = 0.f, tcv = 0.f;
    for (int k = lane; k < DC_; k += 32) {
        const float a0 = xw[k], a1 = xh[k], a2 = xc[k];
        z   += a0 * wg[k] + a1 * wg[DC_ + k] + a2 * wg[2 * DC_ + k];
        s   += a2 * ws[k];
        tcv += a0 * wt[k] + a1 * wt[DC_ + k];
    }
    z   = wave_reduce_sum(z);
    s   = wave_reduce_sum(s);
    tcv = wave_reduce_sum(tcv);
    if (lane == 0) {
        const float zt = 1.f / (1.f + __expf(-(z + bg[d])));
        const float sc = tanhf(s + bs[d]);
        const float tc = tanhf(tcv + bt[d]);
        out[wid] = zt * sc + (1.f - zt) * tc;
    }
}

extern "C" void kernel_launch(void* const* d_in, const int* in_sizes, int n_in,
                              void* d_out, int out_size, void* d_ws, size_t ws_size,
                              hipStream_t stream) {
    const float* cap  = (const float*)d_in[0];
    const float* dh   = (const float*)d_in[1];
    const float* word = (const float*)d_in[2];
    const int*   mask = (const int*)  d_in[3];
    const float* Wf = (const float*)d_in[4];
    const float* bf = (const float*)d_in[5];
    const float* Wd = (const float*)d_in[6];
    const float* bd = (const float*)d_in[7];
    const float* Wa = (const float*)d_in[8];
    const float* ba = (const float*)d_in[9];
    const float* Wg = (const float*)d_in[10];
    const float* bg = (const float*)d_in[11];
    const float* Ws = (const float*)d_in[12];
    const float* bs = (const float*)d_in[13];
    const float* Wt = (const float*)d_in[14];
    const float* bt = (const float*)d_in[15];
    float* out = (float*)d_out;

    char* wsb = (char*)d_ws;
    float*    scores = (float*)(wsb + 0);                 // B*L f32
    float*    u      = (float*)(wsb + 131072);            // B*A f32
    float*    ctx    = (float*)(wsb + 262144);            // B*DC f32
    _Float16* wf16   = (_Float16*)(wsb + 393216);         // A*DC f16 (2MB)
    float*    alpha  = out + B_ * DC_;                    // second output

    conv_wf_kernel<<<(A_ * DC_) / 256, 256, 0, stream>>>(Wf, wf16, A_ * DC_);
    u_kernel<<<(B_ * A_) / 8, 256, 0, stream>>>(dh, Wd, bd, bf, u);
    scores_kernel<<<(B_ * L_ / 16) / 8, 256, 0, stream>>>(cap, wf16, u, Wa, ba, scores);
    softmax_kernel<<<B_, 256, 0, stream>>>(scores, mask, alpha);
    context_kernel<<<dim3(DC_ / 256, B_), 256, 0, stream>>>(alpha, cap, ctx);
    fuse_kernel<<<(B_ * DC_) / 8, 256, 0, stream>>>(word, dh, ctx, Wg, bg, Ws, bs, Wt, bt, out);
}
